// GNN_87058987090015
// MI455X (gfx1250) — matrix operand order repaired and checked by reading, not compiled, not run on hardware
//
#include <hip/hip_runtime.h>
#include <math.h>

typedef __attribute__((ext_vector_type(16))) __bf16 v16bf;
typedef __attribute__((ext_vector_type(8)))  __bf16 v8bf;
typedef __attribute__((ext_vector_type(8)))  float  v8f;

#define F1 128
#define NH1 8
#define D1 16

// ---------- helpers ----------

__device__ __forceinline__ float lrelu(float x) { return x > 0.0f ? x : 0.2f * x; }
__device__ __forceinline__ float eluf(float x)  { return x > 0.0f ? x : expm1f(x); }

// Order-preserving float atomic max (valid for non-NaN values, init = -inf).
__device__ __forceinline__ void atomicMaxF(float* addr, float val) {
    if (val >= 0.0f) {
        atomicMax((int*)addr, __float_as_int(val));
    } else {
        atomicMin((unsigned int*)addr, __float_as_uint(val));
    }
}

__device__ __forceinline__ void edge_sd(const int* __restrict__ ei, int E, int e,
                                        int& s, int& d) {
    if (e < E) { s = ei[e]; d = ei[E + e]; }
    else       { s = e - E; d = s; }            // self loops appended
}

// ---------- init ----------

__global__ void fill_f32(float* __restrict__ p, float v, long long n) {
    long long i = (long long)blockIdx.x * blockDim.x + threadIdx.x;
    if (i < n) p[i] = v;
}

// ---------- W1 pre-transpose + bf16 convert: W1T[n][k] = bf16(W1[k][n]) ----------

__global__ void prep_w1t(const float* __restrict__ W1, __bf16* __restrict__ W1T) {
    const int g = blockIdx.x * blockDim.x + threadIdx.x;
    if (g >= F1 * F1) return;
    const int n = g >> 7, k = g & 127;
    W1T[n * F1 + k] = (__bf16)W1[k * F1 + n];
}

// ---------- layer 1 GEMM: h1[N,128] = x[N,128] @ W1[128,128] (bf16 WMMA) ----------

__global__ void gemm1_wmma(const float* __restrict__ x, const __bf16* __restrict__ W1T,
                           float* __restrict__ h1, int N, int nRowTiles) {
    const int lane = threadIdx.x & 31;
    const int wave = threadIdx.x >> 5;
    const int rt   = blockIdx.x * (blockDim.x >> 5) + wave;
    if (rt >= nRowTiles) return;                 // uniform per-wave exit

    const int n0       = rt * 16;
    const int halfSel  = (lane < 16) ? 0 : 1;    // lane half selects K sub-block
    const int lr       = lane & 15;
    const bool fullTile = (n0 + 16 <= N);        // wave-uniform

    // Preload 4 A fragments (K chunks of 32), ISA 16-bit A 16x32 layout:
    // lanes 0-15:  halves 0..7 = K 0..7,  halves 8..15 = K 16..23 (row M = lane)
    // lanes 16-31: halves 0..7 = K 8..15, halves 8..15 = K 24..31
    int arow = n0 + lr; if (arow >= N) arow = N - 1;
    const float4* xr4 = (const float4*)(x + (long long)arow * F1);
    v16bf afrag[4];
#pragma unroll
    for (int kc = 0; kc < 4; ++kc) {
        const int kb = kc * 32 + halfSel * 8;    // multiple of 8
        const float4 p0 = xr4[(kb >> 2) + 0];    // K kb..kb+3
        const float4 p1 = xr4[(kb >> 2) + 1];    // K kb+4..kb+7
        const float4 p2 = xr4[(kb >> 2) + 4];    // K kb+16..kb+19
        const float4 p3 = xr4[(kb >> 2) + 5];    // K kb+20..kb+23
        v16bf a;
        a[0]  = (__bf16)p0.x; a[1]  = (__bf16)p0.y; a[2]  = (__bf16)p0.z; a[3]  = (__bf16)p0.w;
        a[4]  = (__bf16)p1.x; a[5]  = (__bf16)p1.y; a[6]  = (__bf16)p1.z; a[7]  = (__bf16)p1.w;
        a[8]  = (__bf16)p2.x; a[9]  = (__bf16)p2.y; a[10] = (__bf16)p2.z; a[11] = (__bf16)p2.w;
        a[12] = (__bf16)p3.x; a[13] = (__bf16)p3.y; a[14] = (__bf16)p3.z; a[15] = (__bf16)p3.w;
        afrag[kc] = a;
    }

    // 8 output column tiles of 16
    for (int nt = 0; nt < 8; ++nt) {
        const int col = nt * 16 + lr;
        const __bf16* wcol = W1T + col * F1;     // contiguous K per column
        v8f acc = {0.f, 0.f, 0.f, 0.f, 0.f, 0.f, 0.f, 0.f};
#pragma unroll
        for (int kc = 0; kc < 4; ++kc) {
            // B 32x16 layout: lane holds column N=lr; lanes 0-15 hold K 0..7,16..23,
            // lanes 16-31 hold K 8..15,24..31 of this chunk (matches A).
            const int kr = kc * 32 + halfSel * 8;
            const v8bf b0 = *(const v8bf*)(wcol + kr);
            const v8bf b1 = *(const v8bf*)(wcol + kr + 16);
            v16bf b;
#pragma unroll
            for (int i = 0; i < 8; ++i) { b[i] = b0[i]; b[8 + i] = b1[i]; }
            acc = __builtin_amdgcn_wmma_f32_16x16x32_bf16(
                false, afrag[kc], false, b, (short)0, acc, false, false);
        }
        // C/D layout: VGPR r -> row M=r (lanes 0-15) or M=r+8 (lanes 16-31)
        float* cbase = h1 + (long long)(n0 + halfSel * 8) * F1 + col;
        if (fullTile) {                           // uniform: unconditional stores
#pragma unroll
            for (int r = 0; r < 8; ++r) cbase[(long long)r * F1] = acc[r];
        } else {
#pragma unroll
            for (int r = 0; r < 8; ++r) {
                const int row = n0 + r + halfSel * 8;
                if (row < N) h1[(long long)row * F1 + col] = acc[r];
            }
        }
    }
}

// ---------- layer 1 attention coefficients ----------

__global__ void attn_vec1(const float* __restrict__ h1, const float* __restrict__ a_s,
                          const float* __restrict__ a_d, float* __restrict__ es,
                          float* __restrict__ ed, int N) {
    const int g = blockIdx.x * blockDim.x + threadIdx.x;
    if (g >= N * NH1) return;
    const int n = g >> 3, h = g & 7;
    const float* hp = h1 + (long long)n * F1 + h * D1;
    float s = 0.f, d = 0.f;
#pragma unroll
    for (int k = 0; k < D1; ++k) {
        const float v = hp[k];
        s += v * a_s[h * D1 + k];
        d += v * a_d[h * D1 + k];
    }
    es[g] = s; ed[g] = d;
}

__global__ void edge_max1(const int* __restrict__ ei, int E, int ET,
                          const float* __restrict__ es, const float* __restrict__ ed,
                          float* __restrict__ m) {
    const int g = blockIdx.x * blockDim.x + threadIdx.x;
    if (g >= ET * NH1) return;
    const int e = g >> 3, h = g & 7;
    int s, d; edge_sd(ei, E, e, s, d);
    const float v = lrelu(es[s * NH1 + h] + ed[d * NH1 + h]);
    atomicMaxF(&m[d * NH1 + h], v);
}

// 32 threads/edge; each thread handles 4 contiguous channels (float4 gather).
__global__ void edge_acc1(const int* __restrict__ ei, int E, long long total,
                          const float* __restrict__ es, const float* __restrict__ ed,
                          const float* __restrict__ m, const float* __restrict__ h1,
                          float* __restrict__ denom, float* __restrict__ acc) {
    const long long g = (long long)blockIdx.x * blockDim.x + threadIdx.x;
    if (g >= total) return;
    const int e = (int)(g >> 5);
    const int q = (int)(g & 31);                 // quad index: channels 4q..4q+3
    const int h = q >> 2;
    int s, d; edge_sd(ei, E, e, s, d);
    const float v = lrelu(es[s * NH1 + h] + ed[d * NH1 + h]);
    const float w = __expf(v - m[d * NH1 + h]);
    if ((q & 3) == 0) atomicAdd(&denom[d * NH1 + h], w);
    const float4 hv = *(const float4*)(h1 + (long long)s * F1 + q * 4);
    float* ap = acc + (long long)d * F1 + q * 4;
    atomicAdd(ap + 0, w * hv.x);
    atomicAdd(ap + 1, w * hv.y);
    atomicAdd(ap + 2, w * hv.z);
    atomicAdd(ap + 3, w * hv.w);
}

__global__ void finalize1(float* __restrict__ acc, const float* __restrict__ denom,
                          const float* __restrict__ b1, int N) {
    const int g = blockIdx.x * blockDim.x + threadIdx.x;
    if (g >= N * F1) return;
    const int n = g >> 7, c = g & 127, h = c >> 4;
    const float v = acc[g] / (denom[n * NH1 + h] + 1e-16f) + b1[c];
    acc[g] = eluf(v);                            // acc buffer becomes x1 in-place
}

// ---------- layer 2 (128 -> 2, single head) ----------

__global__ void layer2_node(const float* __restrict__ x1, const float* __restrict__ W2,
                            const float* __restrict__ a_s2, const float* __restrict__ a_d2,
                            float* __restrict__ h2, float* __restrict__ es2,
                            float* __restrict__ ed2, int N) {
    const int n = blockIdx.x * blockDim.x + threadIdx.x;
    if (n >= N) return;
    const float4* xr4 = (const float4*)(x1 + (long long)n * F1);
    float o0 = 0.f, o1 = 0.f;
#pragma unroll
    for (int k4 = 0; k4 < F1 / 4; ++k4) {
        const float4 v = xr4[k4];
        const int k = k4 * 4;
        o0 += v.x * W2[(k + 0) * 2] + v.y * W2[(k + 1) * 2]
            + v.z * W2[(k + 2) * 2] + v.w * W2[(k + 3) * 2];
        o1 += v.x * W2[(k + 0) * 2 + 1] + v.y * W2[(k + 1) * 2 + 1]
            + v.z * W2[(k + 2) * 2 + 1] + v.w * W2[(k + 3) * 2 + 1];
    }
    h2[n * 2 + 0] = o0; h2[n * 2 + 1] = o1;
    es2[n] = o0 * a_s2[0] + o1 * a_s2[1];
    ed2[n] = o0 * a_d2[0] + o1 * a_d2[1];
}

__global__ void edge_max2(const int* __restrict__ ei, int E, int ET,
                          const float* __restrict__ es2, const float* __restrict__ ed2,
                          float* __restrict__ m2) {
    const int e = blockIdx.x * blockDim.x + threadIdx.x;
    if (e >= ET) return;
    int s, d; edge_sd(ei, E, e, s, d);
    atomicMaxF(&m2[d], lrelu(es2[s] + ed2[d]));
}

__global__ void edge_acc2(const int* __restrict__ ei, int E, int ET,
                          const float* __restrict__ es2, const float* __restrict__ ed2,
                          const float* __restrict__ m2, const float* __restrict__ h2,
                          float* __restrict__ denom2, float* __restrict__ acc2) {
    const int e = blockIdx.x * blockDim.x + threadIdx.x;
    if (e >= ET) return;
    int s, d; edge_sd(ei, E, e, s, d);
    const float v = lrelu(es2[s] + ed2[d]);
    const float w = __expf(v - m2[d]);
    atomicAdd(&denom2[d], w);
    atomicAdd(&acc2[d * 2 + 0], w * h2[s * 2 + 0]);
    atomicAdd(&acc2[d * 2 + 1], w * h2[s * 2 + 1]);
}

__global__ void final2(const float* __restrict__ acc2, const float* __restrict__ denom2,
                       const float* __restrict__ b2, float* __restrict__ out, int N) {
    __shared__ float s0[256], s1[256];
    const int n = blockIdx.x * blockDim.x + threadIdx.x;
    float y0 = 0.f, y1 = 0.f;
    if (n < N) {
        const float inv = 1.0f / (denom2[n] + 1e-16f);
        y0 = eluf(acc2[n * 2 + 0] * inv + b2[0]) / (float)N;
        y1 = eluf(acc2[n * 2 + 1] * inv + b2[1]) / (float)N;
    }
    s0[threadIdx.x] = y0; s1[threadIdx.x] = y1;
    __syncthreads();
    for (int off = 128; off > 0; off >>= 1) {
        if (threadIdx.x < off) {
            s0[threadIdx.x] += s0[threadIdx.x + off];
            s1[threadIdx.x] += s1[threadIdx.x + off];
        }
        __syncthreads();
    }
    if (threadIdx.x == 0) {
        atomicAdd(&out[0], s0[0]);
        atomicAdd(&out[1], s1[0]);
    }
}

// ---------- launcher ----------

extern "C" void kernel_launch(void* const* d_in, const int* in_sizes, int n_in,
                              void* d_out, int out_size, void* d_ws, size_t ws_size,
                              hipStream_t stream) {
    const float* x     = (const float*)d_in[0];
    const int*   ei    = (const int*)  d_in[1];
    const float* W1    = (const float*)d_in[2];
    const float* a_s1  = (const float*)d_in[3];
    const float* a_d1  = (const float*)d_in[4];
    const float* b1    = (const float*)d_in[5];
    const float* W2    = (const float*)d_in[6];
    const float* a_s2  = (const float*)d_in[7];
    const float* a_d2  = (const float*)d_in[8];
    const float* b2    = (const float*)d_in[9];
    float*       out   = (float*)d_out;

    const int N  = in_sizes[0] / F1;
    const int E  = in_sizes[1] / 2;
    const int ET = E + N;                        // edges + self loops

    // workspace layout (floats)
    float* ws = (float*)d_ws;
    size_t o = 0;
    float* h1   = ws + o; o += (size_t)N * F1;
    float* es1  = ws + o; o += (size_t)N * NH1;
    float* ed1  = ws + o; o += (size_t)N * NH1;
    float* m1   = ws + o; o += (size_t)N * NH1;
    float* dn1  = ws + o; o += (size_t)N * NH1;
    float* acc1 = ws + o; o += (size_t)N * F1;   // becomes x1 after finalize1
    float* h2   = ws + o; o += (size_t)N * 2;
    float* es2  = ws + o; o += (size_t)N;
    float* ed2  = ws + o; o += (size_t)N;
    float* m2   = ws + o; o += (size_t)N;
    float* dn2  = ws + o; o += (size_t)N;
    float* acc2 = ws + o; o += (size_t)N * 2;
    __bf16* W1T = (__bf16*)(ws + o); o += (size_t)(F1 * F1) / 2;  // 128x128 bf16

    const int B = 256;
    auto blk = [](long long n, int b) { return (unsigned)((n + b - 1) / b); };

    // init accumulators (deterministic every call)
    fill_f32<<<blk((long long)N * NH1, B), B, 0, stream>>>(m1, -INFINITY, (long long)N * NH1);
    fill_f32<<<blk(N, B), B, 0, stream>>>(m2, -INFINITY, N);
    fill_f32<<<blk((long long)N * NH1, B), B, 0, stream>>>(dn1, 0.f, (long long)N * NH1);
    fill_f32<<<blk(N, B), B, 0, stream>>>(dn2, 0.f, N);
    fill_f32<<<blk((long long)N * F1, B), B, 0, stream>>>(acc1, 0.f, (long long)N * F1);
    fill_f32<<<blk((long long)N * 2, B), B, 0, stream>>>(acc2, 0.f, (long long)N * 2);
    fill_f32<<<1, 32, 0, stream>>>(out, 0.f, out_size);

    // layer 1
    prep_w1t<<<blk(F1 * F1, B), B, 0, stream>>>(W1, W1T);
    const int nRowTiles = (N + 15) / 16;
    gemm1_wmma<<<blk(nRowTiles, 8), 256, 0, stream>>>(x, W1T, h1, N, nRowTiles);
    attn_vec1<<<blk((long long)N * NH1, B), B, 0, stream>>>(h1, a_s1, a_d1, es1, ed1, N);
    edge_max1<<<blk((long long)ET * NH1, B), B, 0, stream>>>(ei, E, ET, es1, ed1, m1);
    const long long tot1 = (long long)ET * 32;   // 32 threads per edge
    edge_acc1<<<blk(tot1, B), B, 0, stream>>>(ei, E, tot1, es1, ed1, m1, h1, dn1, acc1);
    finalize1<<<blk((long long)N * F1, B), B, 0, stream>>>(acc1, dn1, b1, N);

    // layer 2
    layer2_node<<<blk(N, B), B, 0, stream>>>(acc1, W2, a_s2, a_d2, h2, es2, ed2, N);
    edge_max2<<<blk(ET, B), B, 0, stream>>>(ei, E, ET, es2, ed2, m2);
    edge_acc2<<<blk(ET, B), B, 0, stream>>>(ei, E, ET, es2, ed2, m2, h2, dn2, acc2);
    final2<<<blk(N, B), B, 0, stream>>>(acc2, dn2, b2, out, N);
}
